// RGGraph_41540923686976
// MI455X (gfx1250) — compile-verified, matrix operand order checked by reading, not run on hardware
//
#include <hip/hip_runtime.h>
#include <hip/hip_bf16.h>

#define N_NODES 50000
#define M_PAD   50048          // 782 * 64, zero-padded rows for guard-free GEMM
#define N_EDGES 800000
#define NGRAPH  256
#define DIN     126
#define HDIM    128
#define KQVS_W  512            // K|Q|V|S concatenated
#define LDS_STRIDE 136         // halves per row (bank-skewed)

typedef _Float16 h16;
typedef _Float16 v16h __attribute__((ext_vector_type(16)));
typedef _Float16 v8h  __attribute__((ext_vector_type(8)));
typedef float    v8f  __attribute__((ext_vector_type(8)));

// ---------------------------------------------------------------------------
// Pack concat(Wk|Wq|Wv|Ws) [Kdim x 128 each] into WMMA B-fragment order:
// Bp[ntile(32)][kstep(4)][lane(32)][16 halves], per-lane contiguous 32B.
// ---------------------------------------------------------------------------
__global__ void pack_weights_kernel(const float* __restrict__ Wk,
                                    const float* __restrict__ Wq,
                                    const float* __restrict__ Wv,
                                    const float* __restrict__ Ws,
                                    int Kdim, h16* __restrict__ Bp) {
  int t = blockIdx.x * blockDim.x + threadIdx.x;
  if (t >= 32 * 4 * 32) return;
  int ntile = t >> 7;
  int ks    = (t >> 5) & 3;
  int lane  = t & 31;
  int colg  = ntile * 16 + (lane & 15);
  const float* W = (colg < 128) ? Wk : (colg < 256) ? Wq : (colg < 384) ? Wv : Ws;
  int col   = colg & 127;
  int kbase = ks * 32 + ((lane >= 16) ? 8 : 0);
  h16* dst = Bp + (size_t)t * 16;
#pragma unroll
  for (int j = 0; j < 16; ++j) {
    int k = kbase + j + ((j >= 8) ? 8 : 0);
    float v = (k < Kdim) ? W[k * HDIM + col] : 0.f;
    dst[j] = (h16)v;
  }
}

// x [N,126] f32 -> act_h [M_PAD,128] f16 (zero-padded in both K and M)
__global__ void convert_x_kernel(const float* __restrict__ x, h16* __restrict__ xh) {
  int idx = blockIdx.x * blockDim.x + threadIdx.x;
  if (idx >= M_PAD * HDIM) return;
  int n = idx >> 7, c = idx & 127;
  float v = (n < N_NODES && c < DIN) ? x[n * DIN + c] : 0.f;
  xh[idx] = (h16)v;
}

// ---------------------------------------------------------------------------
// Fused KQVS GEMM: C[M_PAD,512] = A[M_PAD,128](f16) * Bp(packed) + bias.
// Block = 256 thr = 8 waves; stages a 64x128 A supertile into LDS via
// global_load_async_to_lds_b128 (ASYNCcnt path). Each wave owns one 16-col
// N tile (grid.y*8+wave), holds its 4 B k-step fragments in registers, and
// sweeps 4 M tiles -> 16 v_wmma_f32_16x16x32_f16 per wave, 4x B reuse.
// ---------------------------------------------------------------------------
__global__ __launch_bounds__(256, 2)
void gemm_kqvs_kernel(const h16* __restrict__ A, const h16* __restrict__ Bp,
                      const float* __restrict__ bk, const float* __restrict__ bq,
                      const float* __restrict__ bv, const float* __restrict__ bs,
                      float* __restrict__ C) {
  __shared__ h16 lds[64 * LDS_STRIDE];   // 64 rows x 128 halves (skewed) = 17 KB
  const int m0 = blockIdx.x * 64;

  // Async-stage A supertile: 16 KB, 256 threads x 4 x b128
  {
    int t = threadIdx.x;
#pragma unroll
    for (int i = 0; i < 4; ++i) {
      int c = t + 256 * i;
      int row = c >> 4, chunk = c & 15;
      const void* gp = (const void*)(A + (size_t)(m0 + row) * HDIM + chunk * 8);
      // generic shared pointer low 32 bits == LDS byte offset (aperture rule)
      unsigned lo = (unsigned)(uintptr_t)(&lds[row * LDS_STRIDE + chunk * 8]);
      asm volatile("global_load_async_to_lds_b128 %0, %1, off"
                   :: "v"(lo), "v"(gp) : "memory");
    }
    asm volatile("s_wait_asynccnt 0x0" ::: "memory");
  }
  __syncthreads();

  const int wave  = threadIdx.x >> 5;
  const int lane  = threadIdx.x & 31;
  const int ntile = blockIdx.y * 8 + wave;
  const int arow  = lane & 15;
  const int ahalf = (lane >> 4) ? 8 : 0;

  // Load the wave's 4 B k-step fragments once (per-lane contiguous 32B each)
  v16h bfr[4];
#pragma unroll
  for (int ks = 0; ks < 4; ++ks) {
    const h16* bp = Bp + (size_t)(((ntile * 4 + ks) * 32) + lane) * 16;
    v8h blo = *(const v8h*)(bp);
    v8h bhi = *(const v8h*)(bp + 8);
#pragma unroll
    for (int i = 0; i < 8; ++i) { bfr[ks][i] = blo[i]; bfr[ks][8 + i] = bhi[i]; }
  }

  const int colg = ntile * 16 + (lane & 15);
  const float* bptr = (colg < 128) ? bk : (colg < 256) ? bq : (colg < 384) ? bv : bs;
  const float bias = bptr[colg & 127];

#pragma unroll
  for (int mt = 0; mt < 4; ++mt) {
    v8f acc = {};
#pragma unroll
    for (int ks = 0; ks < 4; ++ks) {
      int r  = mt * 16 + arow;
      int kb = ks * 32 + ahalf;
      v8h alo = *(const v8h*)(&lds[r * LDS_STRIDE + kb]);
      v8h ahi = *(const v8h*)(&lds[r * LDS_STRIDE + kb + 16]);
      v16h a;
#pragma unroll
      for (int i = 0; i < 8; ++i) { a[i] = alo[i]; a[8 + i] = ahi[i]; }
      acc = __builtin_amdgcn_wmma_f32_16x16x32_f16(
          false, a, false, bfr[ks], (short)0, acc, false, false);
    }
    // D layout: VGPR v -> row base + v (+8 for upper half-lanes), col = colg
    const int rbase = m0 + mt * 16 + ((lane >> 4) ? 8 : 0);
#pragma unroll
    for (int v = 0; v < 8; ++v) {
      C[(size_t)(rbase + v) * KQVS_W + colg] = acc[v] + bias;
    }
  }
}

// ---------------------------------------------------------------------------
// Edge message pass: one wave per edge, float4 per lane (128 dims / 32 lanes).
// gate = sigmoid(K[dst]+Q[src]); scatter-add gate*V[src] into agg[dst].
// KQVS/agg working set (~128MB) fits in the 192MB L2 -> gathers mostly L2 hits.
// ---------------------------------------------------------------------------
__global__ void edge_scatter_kernel(const float* __restrict__ KQVS,
                                    const long long* __restrict__ ei,
                                    float* __restrict__ agg) {
  int gid  = blockIdx.x * blockDim.x + threadIdx.x;
  int e    = gid >> 5;
  int lane = gid & 31;
  if (e >= N_EDGES) return;
  int src = (int)ei[e];
  int dst = (int)ei[N_EDGES + e];
  int d = lane * 4;
  float4 k = *(const float4*)(KQVS + (size_t)dst * KQVS_W + 0   + d);
  float4 q = *(const float4*)(KQVS + (size_t)src * KQVS_W + 128 + d);
  float4 v = *(const float4*)(KQVS + (size_t)src * KQVS_W + 256 + d);
  float4 m;
  m.x = v.x / (1.f + __expf(-(k.x + q.x)));
  m.y = v.y / (1.f + __expf(-(k.y + q.y)));
  m.z = v.z / (1.f + __expf(-(k.z + q.z)));
  m.w = v.w / (1.f + __expf(-(k.w + q.w)));
  float* ap = agg + (size_t)dst * HDIM + d;
  atomicAdd(ap + 0, m.x);
  atomicAdd(ap + 1, m.y);
  atomicAdd(ap + 2, m.z);
  atomicAdd(ap + 3, m.w);
}

// h = (relu?)(agg + S). Layers 1-2: write f32 (unused pad rows untouched) and
// f16 activations for the next GEMM. Layer 3: scatter straight into hg pool.
__global__ void finalize_kernel(float* __restrict__ agg, const float* __restrict__ KQVS,
                                h16* __restrict__ act_h, int do_relu, int do_pool,
                                const long long* __restrict__ batch,
                                float* __restrict__ hg) {
  int idx = blockIdx.x * blockDim.x + threadIdx.x;
  if (idx >= N_NODES * HDIM) return;
  int n = idx >> 7, c = idx & 127;
  float v = agg[idx] + KQVS[(size_t)n * KQVS_W + 384 + c];
  if (do_relu) v = fmaxf(v, 0.f);
  if (do_pool) {
    atomicAdd(&hg[(int)batch[n] * HDIM + c], v);
  } else {
    agg[idx]   = v;
    act_h[idx] = (h16)v;
  }
}

// Tiny MLP head: per-graph block of 64 threads. [128]->[64] relu -> [1] sigmoid
__global__ void mlp_kernel(const float* __restrict__ hg, const float* __restrict__ W4,
                           const float* __restrict__ b4, const float* __restrict__ W5,
                           const float* __restrict__ b5, float* __restrict__ out) {
  __shared__ float sh[HDIM];
  __shared__ float s2[64];
  int g = blockIdx.x;
  int t = threadIdx.x;                 // 64 threads
  sh[t]      = hg[g * HDIM + t];
  sh[t + 64] = hg[g * HDIM + t + 64];
  __syncthreads();
  float acc = b4[t];
#pragma unroll 8
  for (int k = 0; k < HDIM; ++k) acc += sh[k] * W4[k * 64 + t];
  s2[t] = fmaxf(acc, 0.f) * W5[t];
  __syncthreads();
  if (t < 32) {
    float p = s2[t] + s2[t + 32];
#pragma unroll
    for (int off = 16; off; off >>= 1) p += __shfl_down(p, off, 32);
    if (t == 0) out[g] = 1.f / (1.f + __expf(-(p + b5[0])));
  }
}

// ---------------------------------------------------------------------------
extern "C" void kernel_launch(void* const* d_in, const int* in_sizes, int n_in,
                              void* d_out, int out_size, void* d_ws, size_t ws_size,
                              hipStream_t stream) {
  (void)in_sizes; (void)n_in; (void)out_size; (void)ws_size;

  const float* x  = (const float*)d_in[0];
  // d_in[1] edge_attr: unused by the conv (edge_dim=None)
  const float* Wk[3] = {(const float*)d_in[2],  (const float*)d_in[10], (const float*)d_in[18]};
  const float* bk[3] = {(const float*)d_in[3],  (const float*)d_in[11], (const float*)d_in[19]};
  const float* Wq[3] = {(const float*)d_in[4],  (const float*)d_in[12], (const float*)d_in[20]};
  const float* bq[3] = {(const float*)d_in[5],  (const float*)d_in[13], (const float*)d_in[21]};
  const float* Wv[3] = {(const float*)d_in[6],  (const float*)d_in[14], (const float*)d_in[22]};
  const float* bv[3] = {(const float*)d_in[7],  (const float*)d_in[15], (const float*)d_in[23]};
  const float* Ws_[3]= {(const float*)d_in[8],  (const float*)d_in[16], (const float*)d_in[24]};
  const float* bs[3] = {(const float*)d_in[9],  (const float*)d_in[17], (const float*)d_in[25]};
  const float* W4 = (const float*)d_in[26];
  const float* b4 = (const float*)d_in[27];
  const float* W5 = (const float*)d_in[28];
  const float* b5 = (const float*)d_in[29];
  const long long* ei    = (const long long*)d_in[30];  // int64 [2,E]
  const long long* batch = (const long long*)d_in[31];  // int64 [N]

  // Workspace carve-up (256B aligned slices)
  uint8_t* ws = (uint8_t*)d_ws;
  size_t off = 0;
  auto alloc = [&](size_t bytes) -> void* {
    void* p = ws + off;
    off = (off + bytes + 255) & ~(size_t)255;
    return p;
  };
  h16*   act_h = (h16*)alloc((size_t)M_PAD * HDIM * sizeof(h16));
  h16*   Bp[3];
  for (int l = 0; l < 3; ++l) Bp[l] = (h16*)alloc((size_t)32 * 4 * 32 * 16 * sizeof(h16));
  float* KQVS  = (float*)alloc((size_t)M_PAD * KQVS_W * sizeof(float));
  float* agg   = (float*)alloc((size_t)N_NODES * HDIM * sizeof(float));
  float* hg    = (float*)alloc((size_t)NGRAPH * HDIM * sizeof(float));

  const int blocksN   = (N_NODES * HDIM + 255) / 256;
  const int blocksPad = (M_PAD * HDIM + 255) / 256;

  // Pack weights (tiny, once per call) + convert input activations to f16
  for (int l = 0; l < 3; ++l)
    pack_weights_kernel<<<16, 256, 0, stream>>>(Wk[l], Wq[l], Wv[l], Ws_[l],
                                                l == 0 ? DIN : HDIM, Bp[l]);
  convert_x_kernel<<<blocksPad, 256, 0, stream>>>(x, act_h);
  hipMemsetAsync(hg, 0, (size_t)NGRAPH * HDIM * sizeof(float), stream);

  for (int l = 0; l < 3; ++l) {
    hipMemsetAsync(agg, 0, (size_t)N_NODES * HDIM * sizeof(float), stream);
    gemm_kqvs_kernel<<<dim3(M_PAD / 64, 4), 256, 0, stream>>>(
        act_h, Bp[l], bk[l], bq[l], bv[l], bs[l], KQVS);
    edge_scatter_kernel<<<(N_EDGES * 32) / 256, 256, 0, stream>>>(KQVS, ei, agg);
    finalize_kernel<<<blocksN, 256, 0, stream>>>(agg, KQVS, act_h,
                                                 /*relu=*/l < 2 ? 1 : 0,
                                                 /*pool=*/l == 2 ? 1 : 0,
                                                 batch, hg);
  }

  mlp_kernel<<<NGRAPH, 64, 0, stream>>>(hg, W4, b4, W5, b5, (float*)d_out);
}
